// MultiHeadAttentionQuantum_65481071399836
// MI455X (gfx1250) — matrix operand order repaired and checked by reading, not compiled, not run on hardware
//
#include <hip/hip_runtime.h>
#include <math.h>
#include <stdint.h>

// Problem constants (reference: B=4, S=2048, E=1024, H=16, DK=64)
constexpr int B_  = 4;
constexpr int S_  = 2048;
constexpr int E_  = 1024;
constexpr int H_  = 16;
constexpr int DK_ = 64;

typedef __attribute__((ext_vector_type(16))) _Float16 v16h;
typedef __attribute__((ext_vector_type(8)))  _Float16 v8h;
typedef __attribute__((ext_vector_type(4)))  _Float16 v4h;
typedef __attribute__((ext_vector_type(8)))  float    v8f;

static __device__ inline v8f wmma16(v16h a, v16h b, v8f c) {
  // D = A(16x32 f16) * B(32x16 f16) + C(16x16 f32)
  return __builtin_amdgcn_wmma_f32_16x16x32_f16(
      /*neg_a=*/false, a, /*neg_b=*/false, b,
      /*c_mod=*/(short)0, c, /*reuse_a=*/false, /*reuse_b=*/false);
}

// ---- CDNA5 async global->LDS staging (tracked by ASYNCcnt) ----------------
// LDS byte address = low 32 bits of the flat shared pointer (ISA 10.2:
// LDS aperture).
static __device__ inline void async_ld_b128(uint32_t lds_addr, const void* g) {
  asm volatile("global_load_async_to_lds_b128 %0, %1, off"
               :: "v"(lds_addr), "v"((uint64_t)(uintptr_t)g)
               : "memory");
}
static __device__ inline void wait_async0() {
#if __has_builtin(__builtin_amdgcn_s_wait_asynccnt)
  __builtin_amdgcn_s_wait_asynccnt(0);
#else
  asm volatile("s_wait_asynccnt 0" ::: "memory");
#endif
}

// ---- A-fragment: 16x32 f16, M = lane&15, K = 16*(lane>>4) + e -------------
static __device__ inline v16h afrag_f16(const _Float16* __restrict__ base,
                                        int stride, int lane) {
  const _Float16* p = base + (size_t)(lane & 15) * stride + ((lane >> 4) << 4);
  v8h lo = *(const v8h*)p;
  v8h hi = *(const v8h*)(p + 8);
  v16h r;
#pragma unroll
  for (int i = 0; i < 8; ++i) { r[i] = lo[i]; r[i + 8] = hi[i]; }
  return r;
}

// ---- B-fragment: 32x16 f16, N = lane&15, K = 16*(lane>>4) + e -------------
static __device__ inline v16h bfrag_f16(const _Float16* __restrict__ base,
                                        int stride, int lane) {
  const _Float16* p = base + (size_t)(lane & 15) * stride + ((lane >> 4) << 4);
  v8h lo = *(const v8h*)p;
  v8h hi = *(const v8h*)(p + 8);
  v16h r;
#pragma unroll
  for (int i = 0; i < 8; ++i) { r[i] = lo[i]; r[i + 8] = hi[i]; }
  return r;
}

// ============================================================================
// Kernel 0: one-shot fp32 -> f16 conversion (x and the four weight matrices).
// Removes all per-tile convert chains from the GEMM hot loops.
// ============================================================================
__global__ __launch_bounds__(256) void k_cvt(const float* __restrict__ src,
                                             _Float16* __restrict__ dst,
                                             int n4) {
  const int i = blockIdx.x * 256 + threadIdx.x;
  if (i < n4) {
    const float4 v = ((const float4*)src)[i];
    v4h o;
    o[0] = (_Float16)v.x; o[1] = (_Float16)v.y;
    o[2] = (_Float16)v.z; o[3] = (_Float16)v.w;
    ((v4h*)dst)[i] = o;
  }
}

// ============================================================================
// Kernel 1: fused Q/K/V projection + quantum transform cos(xW^T + theta).
// One wave computes a 32x64 output tile (one head's column block).
// Q,K stored [B,H,S,DK] f16; V stored transposed [B,H,DK,S] f16.
// ============================================================================
__global__ __launch_bounds__(128) void k_qkv(
    const _Float16* __restrict__ xh, const _Float16* __restrict__ Wqh,
    const _Float16* __restrict__ Wkh, const _Float16* __restrict__ Wvh,
    const float* __restrict__ theta,
    _Float16* __restrict__ Qh, _Float16* __restrict__ Kh,
    _Float16* __restrict__ Vt) {
  const int lane = threadIdx.x & 31;
  const int wid  = (blockIdx.x << 2) + (threadIdx.x >> 5);
  constexpr int TM = (B_ * S_) / 32;  // 256 row tiles
  constexpr int TN = H_;              // 16 heads
  const int sel = wid / (TM * TN);    // 0:Q 1:K 2:V
  const int t   = wid % (TM * TN);
  const int mt  = t / TN;
  const int h   = t % TN;

  const _Float16* W = (sel == 0) ? Wqh : (sel == 1) ? Wkh : Wvh;

  v8f acc[2][4] = {};
  const _Float16* xr0 = xh + (size_t)mt * 32 * E_;
  const _Float16* xr1 = xr0 + (size_t)16 * E_;
  for (int k = 0; k < E_; k += 32) {
    v16h a0 = afrag_f16(xr0 + k, E_, lane);
    v16h a1 = afrag_f16(xr1 + k, E_, lane);
#pragma unroll
    for (int j = 0; j < 4; ++j) {
      // y = x @ W^T  =>  B(k,n) = W[n,k]  (row-major, contiguous k)
      v16h bb = bfrag_f16(W + (size_t)(h * DK_ + j * 16) * E_ + k, E_, lane);
      acc[0][j] = wmma16(a0, bb, acc[0][j]);
      acc[1][j] = wmma16(a1, bb, acc[1][j]);
    }
  }

  const int col = lane & 15;
  const int mb  = (lane >> 4) << 3;
#pragma unroll
  for (int u = 0; u < 2; ++u) {
#pragma unroll
    for (int j = 0; j < 4; ++j) {
      const int d = j * 16 + col;
      const float th = theta[d];
#pragma unroll
      for (int i = 0; i < 8; ++i) {
        const int r = mt * 32 + u * 16 + mb + i;  // row in [0, B*S)
        const int b = r >> 11;                    // S_ == 2048
        const int s = r & (S_ - 1);
        const float val = __cosf(acc[u][j][i] + th);
        const size_t bh = (size_t)b * H_ + h;
        if (sel == 0)      Qh[(bh * S_ + s) * DK_ + d] = (_Float16)val;
        else if (sel == 1) Kh[(bh * S_ + s) * DK_ + d] = (_Float16)val;
        else               Vt[(bh * DK_ + d) * S_ + s] = (_Float16)val;
      }
    }
  }
}

// ============================================================================
// Kernel 2: flash attention. One block owns (b, h, 64 query rows); its 4
// waves each own a 16-row tile. K/V tiles are staged once per block into LDS
// with global_load_async_to_lds_b128 (ASYNCcnt), shared by all 4 waves.
// P (probs) staged through LDS to convert C-layout -> A-layout.
// ============================================================================
__global__ __launch_bounds__(128) void k_attn(
    const _Float16* __restrict__ Qh, const _Float16* __restrict__ Kh,
    const _Float16* __restrict__ Vt, _Float16* __restrict__ AO) {
  __shared__ __align__(16) _Float16 Klds[32 * 64];     // keys x d   (4 KB)
  __shared__ __align__(16) _Float16 Vlds[64 * 32];     // d x keys   (4 KB)
  __shared__ __align__(16) _Float16 Plds[4][16 * 32];  // per-wave P (4 KB)
  const int tid  = threadIdx.x;
  const int lane = tid & 31;
  const int w    = tid >> 5;
  constexpr int QB = S_ / 64;        // 32 query blocks per head
  const int bh = blockIdx.x / QB;    // 0..B*H-1
  const int qt = (blockIdx.x % QB) * 4 + w;
  const int b  = bh / H_;
  const int h  = bh % H_;

  // Preload + pre-scale Q fragments (1/sqrt(DK) = 0.125)
  const _Float16* qbase = Qh + ((size_t)bh * S_ + (size_t)qt * 16) * DK_;
  v16h a0 = afrag_f16(qbase,      DK_, lane);   // d 0..31
  v16h a1 = afrag_f16(qbase + 32, DK_, lane);   // d 32..63
  const _Float16 qs = (_Float16)0.125f;
#pragma unroll
  for (int i = 0; i < 16; ++i) { a0[i] *= qs; a1[i] *= qs; }

  v8f o[4] = {};
  float mrow[8], lrow[8];
#pragma unroll
  for (int i = 0; i < 8; ++i) { mrow[i] = -1.0e30f; lrow[i] = 0.0f; }

  const _Float16* kh = Kh + (size_t)bh * S_ * DK_;
  const _Float16* vt = Vt + (size_t)bh * DK_ * S_;
  _Float16* P = &Plds[w][0];
  const int col = lane & 15;
  const int mb  = (lane >> 4) << 3;

  // Per-thread staging slots: K tile is 4 KB contiguous (32 keys x 64 d);
  // V tile is 64 rows (d) x 32 keys, 64 B per row from transposed V.
  const uint32_t klA = (uint32_t)(uintptr_t)(&Klds[0]) + (uint32_t)tid * 32u;
  const int vrow = tid >> 1, vcol = (tid & 1) * 16;
  const uint32_t vlA =
      (uint32_t)(uintptr_t)(&Vlds[0]) + (uint32_t)(vrow * 32 + vcol) * 2u;

  for (int j0 = 0; j0 < S_; j0 += 32) {
    __syncthreads();  // previous tile fully consumed before overwrite
    const _Float16* kg = kh + (size_t)j0 * DK_ + (size_t)tid * 16;
    async_ld_b128(klA,       kg);
    async_ld_b128(klA + 16u, kg + 8);
    const _Float16* vg = vt + (size_t)vrow * S_ + j0 + vcol;
    async_ld_b128(vlA,       vg);
    async_ld_b128(vlA + 16u, vg + 8);
    wait_async0();
    __syncthreads();  // all waves' async data visible

    // scores[m,n] = sum_d Q[m,d] K[n,d]; B(d,n) = K-row n, contiguous d
    v16h b00 = bfrag_f16(Klds,                 DK_, lane);  // keys 0..15,  d 0..31
    v16h b01 = bfrag_f16(Klds + 32,            DK_, lane);  // keys 0..15,  d 32..63
    v16h b10 = bfrag_f16(Klds + 16 * DK_,      DK_, lane);  // keys 16..31
    v16h b11 = bfrag_f16(Klds + 16 * DK_ + 32, DK_, lane);
    v8f s0 = {}, s1 = {};
    s0 = wmma16(a0, b00, s0);
    s0 = wmma16(a1, b01, s0);
    s1 = wmma16(a0, b10, s1);
    s1 = wmma16(a1, b11, s1);

    // Online softmax over this 16x32 score block. Row i maps to query row
    // mb+i; its 16 columns live in one 16-lane half, so xor-shuffles
    // {8,4,2,1} reduce within each half independently (wave32).
#pragma unroll
    for (int i = 0; i < 8; ++i) {
      float tm = fmaxf(s0[i], s1[i]);
#pragma unroll
      for (int off = 8; off >= 1; off >>= 1)
        tm = fmaxf(tm, __shfl_xor(tm, off, 32));
      const float mnew = fmaxf(mrow[i], tm);
      const float corr = __expf(mrow[i] - mnew);
      mrow[i] = mnew;
      const float p0 = __expf(s0[i] - mnew);
      const float p1 = __expf(s1[i] - mnew);
      float ps = p0 + p1;
#pragma unroll
      for (int off = 8; off >= 1; off >>= 1)
        ps += __shfl_xor(ps, off, 32);
      lrow[i] = lrow[i] * corr + ps;
#pragma unroll
      for (int tt = 0; tt < 4; ++tt) o[tt][i] *= corr;
      const int m = mb + i;
      P[m * 32 + col]      = (_Float16)p0;   // keys j0+0..15
      P[m * 32 + 16 + col] = (_Float16)p1;   // keys j0+16..31
    }

    // P (16x32) as A-fragment; O += P @ V with V from LDS (d x keys).
    v16h pa = afrag_f16(P, 32, lane);
#pragma unroll
    for (int tt = 0; tt < 4; ++tt) {
      v16h vb = bfrag_f16(Vlds + tt * 16 * 32, 32, lane);
      o[tt] = wmma16(pa, vb, o[tt]);
    }
  }

  // Normalize and write attention output as [B,S,E] f16 (head-interleaved).
#pragma unroll
  for (int tt = 0; tt < 4; ++tt) {
#pragma unroll
    for (int i = 0; i < 8; ++i) {
      const int s = qt * 16 + mb + i;
      const float val = o[tt][i] / lrow[i];
      AO[((size_t)b * S_ + s) * E_ + h * DK_ + tt * 16 + col] = (_Float16)val;
    }
  }
}

// ============================================================================
// Kernel 3: output projection  out = AO @ Wo^T  (fp32 output), 32x64 tiles.
// ============================================================================
__global__ __launch_bounds__(128) void k_oproj(
    const _Float16* __restrict__ AO, const _Float16* __restrict__ Woh,
    float* __restrict__ out) {
  const int lane = threadIdx.x & 31;
  const int wid  = (blockIdx.x << 2) + (threadIdx.x >> 5);
  constexpr int TN = E_ / 64;  // 16 column blocks
  const int mt = wid / TN;
  const int nt = wid % TN;

  v8f acc[2][4] = {};
  const _Float16* ar0 = AO + (size_t)mt * 32 * E_;
  const _Float16* ar1 = ar0 + (size_t)16 * E_;
  for (int k = 0; k < E_; k += 32) {
    v16h a0 = afrag_f16(ar0 + k, E_, lane);
    v16h a1 = afrag_f16(ar1 + k, E_, lane);
#pragma unroll
    for (int j = 0; j < 4; ++j) {
      v16h bb = bfrag_f16(Woh + (size_t)(nt * 64 + j * 16) * E_ + k, E_, lane);
      acc[0][j] = wmma16(a0, bb, acc[0][j]);
      acc[1][j] = wmma16(a1, bb, acc[1][j]);
    }
  }

  const int col = lane & 15;
  const int mb  = (lane >> 4) << 3;
#pragma unroll
  for (int u = 0; u < 2; ++u)
#pragma unroll
    for (int j = 0; j < 4; ++j)
#pragma unroll
      for (int i = 0; i < 8; ++i)
        out[(size_t)(mt * 32 + u * 16 + mb + i) * E_ + nt * 64 + j * 16 + col] =
            acc[u][j][i];
}

// ============================================================================
extern "C" void kernel_launch(void* const* d_in, const int* in_sizes, int n_in,
                              void* d_out, int out_size, void* d_ws, size_t ws_size,
                              hipStream_t stream) {
  (void)in_sizes; (void)n_in; (void)out_size; (void)ws_size;
  const float* x     = (const float*)d_in[0];
  const float* Wq    = (const float*)d_in[1];
  const float* Wk    = (const float*)d_in[2];
  const float* Wv    = (const float*)d_in[3];
  const float* Wo    = (const float*)d_in[4];
  const float* theta = (const float*)d_in[5];
  float* out = (float*)d_out;

  // Workspace carve (f16):
  const size_t nX = (size_t)B_ * S_ * E_;  // 8,388,608
  const size_t nW = (size_t)E_ * E_;       // 1,048,576
  _Float16* xh  = (_Float16*)d_ws;         // [B*S, E]
  _Float16* Wqh = xh  + nX;
  _Float16* Wkh = Wqh + nW;
  _Float16* Wvh = Wkh + nW;
  _Float16* Woh = Wvh + nW;
  _Float16* Qh  = Woh + nW;                // [B,H,S,DK]
  _Float16* Kh  = Qh  + nX;                // [B,H,S,DK]
  _Float16* Vt  = Kh  + nX;                // [B,H,DK,S] (transposed)
  _Float16* AO  = Vt  + nX;                // [B,S,E]
  // total: 5*16MB + 4*2MB = 88 MB

  // 0) one-shot fp32 -> f16 conversions
  k_cvt<<<dim3((int)(nX / 4 / 256)), dim3(256), 0, stream>>>(x,  xh,  (int)(nX / 4));
  k_cvt<<<dim3((int)(nW / 4 / 256)), dim3(256), 0, stream>>>(Wq, Wqh, (int)(nW / 4));
  k_cvt<<<dim3((int)(nW / 4 / 256)), dim3(256), 0, stream>>>(Wk, Wkh, (int)(nW / 4));
  k_cvt<<<dim3((int)(nW / 4 / 256)), dim3(256), 0, stream>>>(Wv, Wvh, (int)(nW / 4));
  k_cvt<<<dim3((int)(nW / 4 / 256)), dim3(256), 0, stream>>>(Wo, Woh, (int)(nW / 4));

  // 1) QKV projection + cos(.+theta): 3 * 256 * 16 waves / 4 per block
  k_qkv<<<dim3((3 * (B_ * S_ / 32) * H_) / 4), dim3(128), 0, stream>>>(
      xh, Wqh, Wkh, Wvh, theta, Qh, Kh, Vt);
  // 2) flash attention: one block per (b, h, 64 query rows)
  k_attn<<<dim3(B_ * H_ * (S_ / 64)), dim3(128), 0, stream>>>(Qh, Kh, Vt, AO);
  // 3) output projection: 256 * 16 waves / 4 per block
  k_oproj<<<dim3(((B_ * S_ / 32) * H_) / 4), dim3(128), 0, stream>>>(
      AO, Woh, out);
}
